// PairwiseScorer_31894427140391
// MI455X (gfx1250) — compile-verified, matrix operand order checked by reading, not run on hardware
//
#include <hip/hip_runtime.h>
#include <stdint.h>

typedef __attribute__((ext_vector_type(16))) __bf16 v16bf;
typedef __attribute__((ext_vector_type(8)))  float  v8f;

#define D_SPAN   512
#define IN_DIM   1596
#define K1PAD    1600
#define HIDDEN   150
#define NPAD     160
#define KS1      50     // K-steps of 32 for GEMM1 (1600/32)
#define KS2      5      // K-steps of 32 for GEMM2 (160/32)
#define NT       10     // N tiles of 16 (160/16)
#define TPB      (NT*32)
#define PAIRS    16
#define PADV     1000.0f

__device__ __forceinline__ unsigned short f2bf(float f) {
  union { float f; unsigned u; } c; c.f = f;
  unsigned u = c.u;
  u += 0x7FFFu + ((u >> 16) & 1u);   // round-to-nearest-even
  return (unsigned short)(u >> 16);
}

// Within a 32-wide K group, map k -> storage position so that the 16 values a
// lane needs (half = lane>>4) are contiguous in fragment-element order:
// frag elem i = g*8+j holds k = g*16 + half*8 + j.
__device__ __forceinline__ int permute32(int kk) {
  int half = (kk >> 3) & 1, g = kk >> 4, j = kk & 7;
  return half * 16 + g * 8 + j;
}

union V16 { uint4 q[2]; v16bf v; };

__device__ __forceinline__ void ldA(V16& a, const uint4* abase, int ks) {
  a.q[0] = abase[ks * 4];      // 64 B (= one 32-K step) per ks, immediate offsets
  a.q[1] = abase[ks * 4 + 1];
}
__device__ __forceinline__ void ldB(V16& b, const uint4* bbase, int ks) {
  b.q[0] = bbase[ks * 64];     // 1024 B per ks, immediate offsets
  b.q[1] = bbase[ks * 64 + 1];
}

// ---------------------------------------------------------------------------
// Repack W1/W2 into bf16 B-fragments, layout [nt][ks][lane][i] so the GEMM
// wave's base pointer is fixed and the ks stride is an immediate:
// idx = (((nt*KS + ks)*32 + lane)*16 + i),
// k = ks*32 + (i>>3)*16 + (lane>>4)*8 + (i&7), n = nt*16 + (lane&15).
// ---------------------------------------------------------------------------
__global__ void prep_w_kernel(const float* __restrict__ W1, const float* __restrict__ W2,
                              unsigned short* __restrict__ B1, unsigned short* __restrict__ B2) {
  int idx = blockIdx.x * 256 + threadIdx.x;
  const int total1 = KS1 * NT * 32 * 16;   // 256000
  const int total2 = KS2 * NT * 32 * 16;   // 25600
  if (idx < total1) {
    int i = idx & 15, lane = (idx >> 4) & 31, tile = idx >> 9;
    int ks = tile % KS1, nt = tile / KS1;
    int half = lane >> 4, n = nt * 16 + (lane & 15);
    int k = ks * 32 + (i >> 3) * 16 + half * 8 + (i & 7);
    float v = (k < IN_DIM && n < HIDDEN) ? W1[k * HIDDEN + n] : 0.f;
    B1[idx] = f2bf(v);
  } else if (idx < total1 + total2) {
    int id2 = idx - total1;
    int i = id2 & 15, lane = (id2 >> 4) & 31, tile = id2 >> 9;
    int ks = tile % KS2, nt = tile / KS2;
    int half = lane >> 4, n = nt * 16 + (lane & 15);
    int k = ks * 32 + (i >> 3) * 16 + half * 8 + (i & 7);
    float v = (k < HIDDEN && n < HIDDEN) ? W2[k * HIDDEN + n] : 0.f;
    B2[id2] = f2bf(v);
  }
}

__global__ void prep_offsets_kernel(const int* __restrict__ group_sizes,
                                    int* __restrict__ offsets, int M) {
  if (blockIdx.x == 0 && threadIdx.x == 0) {
    int acc = 0;
    for (int i = 0; i < M; ++i) { offsets[i] = acc; acc += group_sizes[i]; }
  }
}

// ---------------------------------------------------------------------------
// Fused pair-feature gather + 3-layer MLP. 16 pairs per block, 10 waves each
// owning one 16-wide N tile of the 160-padded hidden dimension.
// ---------------------------------------------------------------------------
__global__ void __launch_bounds__(TPB)
pair_mlp_kernel(const float* __restrict__ span_reps, const float* __restrict__ mention_scores,
                const float* __restrict__ dist_emb, const float* __restrict__ genre_emb,
                const float* __restrict__ speaker_emb,
                const float* __restrict__ b1, const float* __restrict__ b2,
                const float* __restrict__ W3, const float* __restrict__ b3,
                const int* __restrict__ distances, const int* __restrict__ genre_ids,
                const int* __restrict__ speaker_ids, const int* __restrict__ mention_ids,
                const int* __restrict__ antecedent_ids,
                const unsigned short* __restrict__ B1f, const unsigned short* __restrict__ B2f,
                float* __restrict__ scores, int P)
{
  __shared__ __align__(16) unsigned short s_feat[PAIRS * K1PAD]; // 51200 B (bf16, permuted)
  __shared__ __align__(16) unsigned short s_h1[PAIRS * NPAD];    // 5120 B  (bf16, permuted)
  __shared__ int s_mid[PAIRS], s_aid[PAIRS], s_bkt[PAIRS], s_gid[PAIRS], s_sid[PAIRS];
  float* s_h2 = (float*)s_feat;  // overlay: features dead after GEMM1

  const int tid  = threadIdx.x;
  const int base = blockIdx.x * PAIRS;

  if (tid < PAIRS) {
    int pr = base + tid;
    int mid = 0, aid = 0, d = 0, g = 0, s = 0;
    if (pr < P) {
      mid = mention_ids[pr]; aid = antecedent_ids[pr];
      d = distances[pr]; g = genre_ids[pr]; s = speaker_ids[pr];
    }
    int b = (d >= 1) + (d >= 2) + (d >= 3) + (d >= 4) +
            (d >= 8) + (d >= 16) + (d >= 32) + (d >= 64);
    s_mid[tid] = mid; s_aid[tid] = aid; s_bkt[tid] = b; s_gid[tid] = g; s_sid[tid] = s;
  }
  __syncthreads();

  // Stage 16 pair feature rows into LDS as bf16, 8 elements (one contiguous
  // fragment run) per step: float4 x2 loads -> bf16 pack -> one ds_store_b128.
  {
    const int G8 = K1PAD / 8;            // 200 groups per pair
    for (int t = tid; t < PAIRS * G8; t += TPB) {
      int m   = t / G8;
      int k8  = (t - m * G8) * 8;
      const float* mrow = span_reps + (size_t)s_mid[m] * D_SPAN;
      const float* arow = span_reps + (size_t)s_aid[m] * D_SPAN;
      float v[8];
      if (k8 < D_SPAN) {
        const float4* p = (const float4*)(mrow + k8);
        float4 x = p[0], y = p[1];
        v[0]=x.x; v[1]=x.y; v[2]=x.z; v[3]=x.w; v[4]=y.x; v[5]=y.y; v[6]=y.z; v[7]=y.w;
      } else if (k8 < 2 * D_SPAN) {
        const float4* p = (const float4*)(arow + (k8 - D_SPAN));
        float4 x = p[0], y = p[1];
        v[0]=x.x; v[1]=x.y; v[2]=x.z; v[3]=x.w; v[4]=y.x; v[5]=y.y; v[6]=y.z; v[7]=y.w;
      } else if (k8 < 3 * D_SPAN) {
        int kk = k8 - 2 * D_SPAN;
        const float4* pm = (const float4*)(mrow + kk);
        const float4* pa = (const float4*)(arow + kk);
        float4 xm = pm[0], ym = pm[1], xa = pa[0], ya = pa[1];
        v[0]=xm.x*xa.x; v[1]=xm.y*xa.y; v[2]=xm.z*xa.z; v[3]=xm.w*xa.w;
        v[4]=ym.x*ya.x; v[5]=ym.y*ya.y; v[6]=ym.z*ya.z; v[7]=ym.w*ya.w;
      } else {
        for (int e = 0; e < 8; ++e) {
          int k = k8 + e;
          float x;
          if (k < 1556)        x = dist_emb[s_bkt[m] * 20 + (k - 1536)];
          else if (k < 1576)   x = genre_emb[s_gid[m] * 20 + (k - 1556)];
          else if (k < IN_DIM) x = speaker_emb[s_sid[m] * 20 + (k - 1576)];
          else                 x = 0.f;
          v[e] = x;
        }
      }
      unsigned w0 = (unsigned)f2bf(v[0]) | ((unsigned)f2bf(v[1]) << 16);
      unsigned w1 = (unsigned)f2bf(v[2]) | ((unsigned)f2bf(v[3]) << 16);
      unsigned w2 = (unsigned)f2bf(v[4]) | ((unsigned)f2bf(v[5]) << 16);
      unsigned w3 = (unsigned)f2bf(v[6]) | ((unsigned)f2bf(v[7]) << 16);
      uint4 pkt; pkt.x = w0; pkt.y = w1; pkt.z = w2; pkt.w = w3;
      int kk  = k8 & 31;
      int pos = (k8 & ~31) + ((kk >> 3) & 1) * 16 + (kk >> 4) * 8;  // permuted run start
      *(uint4*)(s_feat + m * K1PAD + pos) = pkt;
    }
  }
  __syncthreads();

  const int wave  = tid >> 5;       // N tile
  const int lane  = tid & 31;
  const int mlow  = lane & 15;      // A row / C column-within-tile
  const int half  = lane >> 4;

  // ---- GEMM1: h1 = relu(feat @ W1 + b1), K = 1600 ----
  const uint4* abase1 = (const uint4*)(s_feat + mlow * K1PAD + half * 16);
  const uint4* bbase1 = (const uint4*)B1f + ((size_t)wave * KS1 * 32 + lane) * 2;
  v8f acc = {};
  {
    V16 a0, b0, a1, b1v;
    ldA(a0, abase1, 0); ldB(b0, bbase1, 0);
    #pragma unroll
    for (int ks = 0; ks < KS1; ++ks) {
      if (ks + 1 < KS1) {
        if (ks & 1) { ldA(a0, abase1, ks + 1); ldB(b0, bbase1, ks + 1); }
        else        { ldA(a1, abase1, ks + 1); ldB(b1v, bbase1, ks + 1); }
      }
      const V16& a = (ks & 1) ? a1 : a0;
      const V16& b = (ks & 1) ? b1v : b0;
      acc = __builtin_amdgcn_wmma_f32_16x16x32_bf16(false, a.v, false, b.v,
                                                    (short)0, acc, false, false);
    }
  }
  {
    int n = wave * 16 + mlow;                       // C layout: N = lane&15
    float bias = (n < HIDDEN) ? b1[n] : 0.f;
    int grp = n >> 5, p = permute32(n & 31);
    #pragma unroll
    for (int r = 0; r < 8; ++r) {
      int m = half * 8 + r;                          // C layout: M = (lane>=16)*8 + vgpr
      float h = (n < HIDDEN) ? fmaxf(acc[r] + bias, 0.f) : 0.f;
      s_h1[m * NPAD + grp * 32 + p] = f2bf(h);
    }
  }
  __syncthreads();

  // ---- GEMM2: h2 = relu(h1 @ W2 + b2), K = 160 ----
  const uint4* abase2 = (const uint4*)(s_h1 + mlow * NPAD + half * 16);
  const uint4* bbase2 = (const uint4*)B2f + ((size_t)wave * KS2 * 32 + lane) * 2;
  v8f acc2 = {};
  {
    V16 a0, b0, a1, b1v;
    ldA(a0, abase2, 0); ldB(b0, bbase2, 0);
    #pragma unroll
    for (int ks = 0; ks < KS2; ++ks) {
      if (ks + 1 < KS2) {
        if (ks & 1) { ldA(a0, abase2, ks + 1); ldB(b0, bbase2, ks + 1); }
        else        { ldA(a1, abase2, ks + 1); ldB(b1v, bbase2, ks + 1); }
      }
      const V16& a = (ks & 1) ? a1 : a0;
      const V16& b = (ks & 1) ? b1v : b0;
      acc2 = __builtin_amdgcn_wmma_f32_16x16x32_bf16(false, a.v, false, b.v,
                                                     (short)0, acc2, false, false);
    }
  }
  {
    int n = wave * 16 + mlow;
    float bias = (n < HIDDEN) ? b2[n] : 0.f;
    #pragma unroll
    for (int r = 0; r < 8; ++r) {
      int m = half * 8 + r;
      float h = (n < HIDDEN) ? fmaxf(acc2[r] + bias, 0.f) : 0.f;
      s_h2[m * NPAD + n] = h;
    }
  }
  __syncthreads();

  // ---- Final: am_s = h2 @ W3 + b3 ; score = am_s + ms[mid] + ms[aid] ----
  if (tid < 128) {
    int m = tid >> 3, j = tid & 7;
    float sum = 0.f;
    for (int k = j; k < HIDDEN; k += 8) sum += s_h2[m * NPAD + k] * W3[k];
    sum += __shfl_xor(sum, 1, 32);
    sum += __shfl_xor(sum, 2, 32);
    sum += __shfl_xor(sum, 4, 32);
    if (j == 0) {
      int pr = base + m;
      if (pr < P) {
        float sc = sum + b3[0] + mention_scores[s_mid[m]] + mention_scores[s_aid[m]];
        scores[pr] = sc;
      }
    }
  }
}

// ---------------------------------------------------------------------------
// Ragged softmax per mention group (pairs are contiguous per group) with the
// extra epsilon logit = 0. One 128-thread block per mention row of the output.
// ---------------------------------------------------------------------------
__global__ void __launch_bounds__(128)
seg_softmax_kernel(const float* __restrict__ scores, const int* __restrict__ group_sizes,
                   const int* __restrict__ offsets, float* __restrict__ out, int width) {
  __shared__ float red[128];
  int i = blockIdx.x, t = threadIdx.x;
  int size = group_sizes[i];
  int off  = offsets[i];
  float x = (t < size) ? scores[off + t] : -3.0e38f;
  red[t] = x; __syncthreads();
  for (int s = 64; s > 0; s >>= 1) {
    if (t < s) red[t] = fmaxf(red[t], red[t + s]);
    __syncthreads();
  }
  float mx = fmaxf(red[0], 0.f);          // epsilon logit participates
  __syncthreads();
  float e = (t < size) ? expf(x - mx) : 0.f;
  red[t] = e; __syncthreads();
  for (int s = 64; s > 0; s >>= 1) {
    if (t < s) red[t] += red[t + s];
    __syncthreads();
  }
  float eps_e = expf(-mx);
  float inv = 1.f / (red[0] + eps_e);
  float* row = out + (size_t)i * width;
  if (t < size)            row[t] = e * inv;
  else if (t == size)      row[t] = eps_e * inv;
  else                     row[t] = PADV;
  if (t == 0 && width > 128) {            // position 128 (size <= 128)
    if (size == 128)       row[128] = eps_e * inv;
    else                   row[128] = PADV;
  }
}

extern "C" void kernel_launch(void* const* d_in, const int* in_sizes, int n_in,
                              void* d_out, int out_size, void* d_ws, size_t ws_size,
                              hipStream_t stream) {
  const float* span_reps      = (const float*)d_in[0];
  const float* mention_scores = (const float*)d_in[1];
  const float* dist_emb       = (const float*)d_in[2];
  const float* genre_emb      = (const float*)d_in[3];
  const float* speaker_emb    = (const float*)d_in[4];
  const float* W1             = (const float*)d_in[5];
  const float* b1             = (const float*)d_in[6];
  const float* W2             = (const float*)d_in[7];
  const float* b2             = (const float*)d_in[8];
  const float* W3             = (const float*)d_in[9];
  const float* b3             = (const float*)d_in[10];
  const int* distances        = (const int*)d_in[11];
  const int* genre_ids        = (const int*)d_in[12];
  const int* speaker_ids      = (const int*)d_in[13];
  const int* mention_ids      = (const int*)d_in[14];
  const int* antecedent_ids   = (const int*)d_in[15];
  const int* group_sizes      = (const int*)d_in[18];

  const int P = in_sizes[11];
  const int M = in_sizes[18];
  const int width = out_size / M;   // 129

  // Workspace layout
  char* ws = (char*)d_ws;
  unsigned short* B1f = (unsigned short*)ws;                          // 512000 B
  unsigned short* B2f = (unsigned short*)(ws + 512000);               // 51200 B
  float* scores       = (float*)(ws + 563200);                        // P*4 B
  size_t score_bytes  = (((size_t)P * 4) + 255) & ~(size_t)255;
  int* offsets        = (int*)(ws + 563200 + score_bytes);            // M*4 B

  const int totalW = KS1 * NT * 32 * 16 + KS2 * NT * 32 * 16;
  prep_w_kernel<<<(totalW + 255) / 256, 256, 0, stream>>>(W1, W2, B1f, B2f);
  prep_offsets_kernel<<<1, 1, 0, stream>>>(group_sizes, offsets, M);

  const int nblk = (P + PAIRS - 1) / PAIRS;
  pair_mlp_kernel<<<nblk, TPB, 0, stream>>>(
      span_reps, mention_scores, dist_emb, genre_emb, speaker_emb,
      b1, b2, W3, b3, distances, genre_ids, speaker_ids,
      mention_ids, antecedent_ids, B1f, B2f, scores, P);

  seg_softmax_kernel<<<M, 128, 0, stream>>>(scores, group_sizes, offsets,
                                            (float*)d_out, width);
}
